// FlowDenoisingLayer_53137335386466
// MI455X (gfx1250) — matrix operand order repaired and checked by reading, not compiled, not run on hardware
//
#include <hip/hip_runtime.h>
#include <hip/hip_bf16.h>
#include <math.h>

typedef __bf16 bf16;
typedef __attribute__((ext_vector_type(16))) __bf16 bf16x16;
typedef __attribute__((ext_vector_type(8)))  __bf16 bf16x8;
typedef __attribute__((ext_vector_type(8)))  float  f32x8;

#define H_DIM   512
#define N_HEADS 8
#define HEAD_D  64
#define BATCH   8
#define ASEQ    1024
#define NV      1024

// ---------------------------------------------------------------------------
// WMMA helpers
// ---------------------------------------------------------------------------
__device__ __forceinline__ f32x8 wmma_bf16(bf16x16 a, bf16x16 b, f32x8 c) {
  // D = A(16x32) * B(32x16) + C, fp32 accumulate
  return __builtin_amdgcn_wmma_f32_16x16x32_bf16(false, a, false, b, (short)0, c,
                                                 false, false);
}

// A-operand fragment (16x32, 16-bit): lane = M row (l&15), half (l>>4) selects
// K in two contiguous 8-element runs at h*8 and 16+h*8 (per ISA A-matrix table).
__device__ __forceinline__ bf16x16 load_frag_a(const bf16* p, int ld) {
  const int lane = threadIdx.x & 31;
  const int r = lane & 15, h = lane >> 4;
  const bf16* q = p + (size_t)r * ld + h * 8;
  bf16x8 lo = *(const bf16x8*)(q);
  bf16x8 hi = *(const bf16x8*)(q + 16);
  bf16x16 out;
#pragma unroll
  for (int i = 0; i < 8; ++i) { out[i] = lo[i]; out[8 + i] = hi[i]; }
  return out;
}

// B-operand fragment (32x16, 16-bit): lane = N col (l&15), half (l>>4) selects
// K in two contiguous 8-element runs at h*8 and 16+h*8 (matches A). 'ld' is the
// stride between the 16 B columns in memory; K runs contiguously.
__device__ __forceinline__ bf16x16 load_frag_b(const bf16* p, int ld) {
  const int lane = threadIdx.x & 31;
  const int n = lane & 15, h = lane >> 4;
  const bf16* q = p + (size_t)n * ld + h * 8;
  bf16x8 lo = *(const bf16x8*)(q);
  bf16x8 hi = *(const bf16x8*)(q + 16);
  bf16x16 out;
#pragma unroll
  for (int i = 0; i < 8; ++i) { out[i] = lo[i]; out[8 + i] = hi[i]; }
  return out;
}

// ---------------------------------------------------------------------------
// fp32 -> bf16 conversion
// ---------------------------------------------------------------------------
__global__ void cvt_f32_bf16(const float* __restrict__ in, bf16* __restrict__ out,
                             int n) {
  int i = blockIdx.x * blockDim.x + threadIdx.x;
  if (i < n) out[i] = (bf16)in[i];
}

// ---------------------------------------------------------------------------
// Generic GEMM:  C[M,N] = A[M,K](bf16,row) * W[N,K](bf16,row)^T + bias
// act==1 -> exact GELU.  Writes f32 (Cf) and/or bf16 (Cb).
// Block = 256 thr (8 waves), wave tile 32x64, WG tile 128x128.
// ---------------------------------------------------------------------------
__global__ __launch_bounds__(256)
void gemm_bf16_wmma(const bf16* __restrict__ A, int lda,
                    const bf16* __restrict__ W, int ldw,
                    const float* __restrict__ bias,
                    float* __restrict__ Cf, bf16* __restrict__ Cb, int ldc,
                    int K, int act)
{
  const int wave = threadIdx.x >> 5;
  const int lane = threadIdx.x & 31;
  const int l15 = lane & 15, hh = lane >> 4;
  const int wm = wave >> 1, wn = wave & 1;
  const int mbase = blockIdx.y * 128 + wm * 32;
  const int nbase = blockIdx.x * 128 + wn * 64;

  f32x8 acc[2][4];
#pragma unroll
  for (int i = 0; i < 2; ++i)
#pragma unroll
    for (int j = 0; j < 4; ++j)
#pragma unroll
      for (int r = 0; r < 8; ++r) acc[i][j][r] = 0.0f;

  const bf16* a0p = A + (size_t)mbase * lda;
  const bf16* a1p = A + (size_t)(mbase + 16) * lda;

  for (int k = 0; k < K; k += 32) {
    __builtin_prefetch(a0p + k + 128, 0, 0);
    bf16x16 a0 = load_frag_a(a0p + k, lda);
    bf16x16 a1 = load_frag_a(a1p + k, lda);
#pragma unroll
    for (int j = 0; j < 4; ++j) {
      bf16x16 bj = load_frag_b(W + (size_t)(nbase + j * 16) * ldw + k, ldw);
      acc[0][j] = wmma_bf16(a0, bj, acc[0][j]);
      acc[1][j] = wmma_bf16(a1, bj, acc[1][j]);
    }
  }

#pragma unroll
  for (int i = 0; i < 2; ++i)
#pragma unroll
    for (int j = 0; j < 4; ++j) {
      const int col = nbase + j * 16 + l15;
      const float bv = bias ? bias[col] : 0.0f;
#pragma unroll
      for (int r = 0; r < 8; ++r) {
        const int row = mbase + i * 16 + r + 8 * hh;
        float v = acc[i][j][r] + bv;
        if (act == 1) v = 0.5f * v * (1.0f + erff(v * 0.70710678118654752f));
        const size_t o = (size_t)row * ldc + col;
        if (Cf) Cf[o] = v;
        if (Cb) Cb[o] = (bf16)v;
      }
    }
}

// ---------------------------------------------------------------------------
// Build transposed V buffers (bf16) so P@V B-fragments load contiguously.
// vt[((b*8+h)*64+d)*1024 + key] = qkv[(b*1024+key)*1536 + 1024 + h*64 + d]
// ---------------------------------------------------------------------------
__global__ void build_vt_self(const bf16* __restrict__ qkvb, bf16* __restrict__ vt) {
  int idx = blockIdx.x * blockDim.x + threadIdx.x;
  if (idx >= BATCH * N_HEADS * HEAD_D * ASEQ) return;
  int key = idx & (ASEQ - 1);
  int t = idx >> 10;
  int d = t & (HEAD_D - 1); t >>= 6;
  int h = t & (N_HEADS - 1);
  int b = t >> 3;
  vt[idx] = qkvb[((size_t)(b * ASEQ + key)) * 1536 + 1024 + h * HEAD_D + d];
}

// cavt[(h*64+d)*1024 + key] = cakv[key*1024 + 512 + h*64 + d]
__global__ void build_vt_cross(const bf16* __restrict__ cakvb, bf16* __restrict__ cavt) {
  int idx = blockIdx.x * blockDim.x + threadIdx.x;
  if (idx >= N_HEADS * HEAD_D * NV) return;
  int key = idx & (NV - 1);
  int hd = idx >> 10;  // 0..511
  cavt[idx] = cakvb[(size_t)key * 1024 + 512 + hd];
}

// ---------------------------------------------------------------------------
// Flash-style attention, one wave per 16 query rows per (b,h).
// mode 0: self-attn over atoms, key mask = target_mask[b][key]
// mode 1: cross-attn over 1024 virtual keys, key mask = (vbi[key]==b)
// Writes O bf16 [B*A, 512] (heads concatenated).
// ---------------------------------------------------------------------------
__global__ __launch_bounds__(128)
void attn_wmma(int mode,
               const bf16* __restrict__ Qbuf,
               const bf16* __restrict__ Kbuf,
               const bf16* __restrict__ Vtbuf,
               const unsigned char* __restrict__ amask,
               const int* __restrict__ vbi,
               bf16* __restrict__ O)
{
  __shared__ __align__(16) bf16 plds[4][16][32];
  const int bh = blockIdx.x;
  const int b = bh >> 3, h = bh & 7;
  const int wave = threadIdx.x >> 5;
  const int lane = threadIdx.x & 31;
  const int l15 = lane & 15, hh = lane >> 4;
  const int qrow0 = blockIdx.y * 64 + wave * 16;

  const bf16 *Qp, *Kp, *Vtp;
  int ldq, ldk;
  if (mode == 0) {
    ldq = 3 * H_DIM;
    Qp  = Qbuf + ((size_t)(b * ASEQ + qrow0)) * ldq + h * HEAD_D;
    Kp  = Kbuf + ((size_t)(b * ASEQ)) * ldq + H_DIM + h * HEAD_D;
    ldk = ldq;
    Vtp = Vtbuf + ((size_t)(b * N_HEADS + h) * HEAD_D) * ASEQ;
  } else {
    ldq = H_DIM;
    Qp  = Qbuf + ((size_t)(b * ASEQ + qrow0)) * ldq + h * HEAD_D;
    Kp  = Kbuf + h * HEAD_D;
    ldk = 2 * H_DIM;
    Vtp = Vtbuf + ((size_t)h * HEAD_D) * NV;
  }

  const bf16x16 aq0 = load_frag_a(Qp, ldq);        // d = 0..31
  const bf16x16 aq1 = load_frag_a(Qp + 32, ldq);   // d = 32..63

  f32x8 o0, o1, o2, o3;
#pragma unroll
  for (int r = 0; r < 8; ++r) { o0[r] = 0; o1[r] = 0; o2[r] = 0; o3[r] = 0; }
  float m[8], lsum[8];
#pragma unroll
  for (int r = 0; r < 8; ++r) { m[r] = -3.0e38f; lsum[r] = 0.0f; }

  for (int kb = 0; kb < NV; kb += 32) {
    f32x8 s0, s1;
#pragma unroll
    for (int r = 0; r < 8; ++r) { s0[r] = 0; s1[r] = 0; }
    const bf16* k0 = Kp + (size_t)kb * ldk;
    const bf16* k1 = Kp + (size_t)(kb + 16) * ldk;
    s0 = wmma_bf16(aq0, load_frag_b(k0,      ldk), s0);
    s0 = wmma_bf16(aq1, load_frag_b(k0 + 32, ldk), s0);
    s1 = wmma_bf16(aq0, load_frag_b(k1,      ldk), s1);
    s1 = wmma_bf16(aq1, load_frag_b(k1 + 32, ldk), s1);

    const int key0 = kb + l15, key1 = kb + 16 + l15;
    bool va, vb2;
    if (mode == 0) {
      va  = amask[(size_t)b * ASEQ + key0] != 0;
      vb2 = amask[(size_t)b * ASEQ + key1] != 0;
    } else {
      va  = (vbi[key0] == b);
      vb2 = (vbi[key1] == b);
    }

#pragma unroll
    for (int r = 0; r < 8; ++r) {
      float x0 = va  ? s0[r] * 0.125f : -1.0e9f;  // 1/sqrt(64)
      float x1 = vb2 ? s1[r] * 0.125f : -1.0e9f;
      float t = fmaxf(x0, x1);
      t = fmaxf(t, __shfl_xor(t, 1, 32));
      t = fmaxf(t, __shfl_xor(t, 2, 32));
      t = fmaxf(t, __shfl_xor(t, 4, 32));
      t = fmaxf(t, __shfl_xor(t, 8, 32));        // row-max within 16-lane half
      const float nm = fmaxf(m[r], t);
      const float corr = __expf(m[r] - nm);
      m[r] = nm;
      const float e0 = __expf(x0 - nm);
      const float e1 = __expf(x1 - nm);
      float rs = e0 + e1;
      rs += __shfl_xor(rs, 1, 32);
      rs += __shfl_xor(rs, 2, 32);
      rs += __shfl_xor(rs, 4, 32);
      rs += __shfl_xor(rs, 8, 32);
      lsum[r] = lsum[r] * corr + rs;
      o0[r] *= corr; o1[r] *= corr; o2[r] *= corr; o3[r] *= corr;
      // stash P (C-layout) into LDS for re-striping into A-layout
      plds[wave][r + 8 * hh][l15]      = (bf16)e0;
      plds[wave][r + 8 * hh][16 + l15] = (bf16)e1;
    }
    // LDS ops from one wave complete in order; wait for all stores, block
    // compiler reordering across the re-striping load.
    asm volatile("s_wait_dscnt 0" ::: "memory");
    const bf16x16 ap = load_frag_a((const bf16*)&plds[wave][0][0], 32);
    asm volatile("" ::: "memory");

    const bf16* vp = Vtp + kb;   // Vt[d][key], ld = 1024 keys
    o0 = wmma_bf16(ap, load_frag_b(vp + (size_t)0  * 1024, 1024), o0);
    o1 = wmma_bf16(ap, load_frag_b(vp + (size_t)16 * 1024, 1024), o1);
    o2 = wmma_bf16(ap, load_frag_b(vp + (size_t)32 * 1024, 1024), o2);
    o3 = wmma_bf16(ap, load_frag_b(vp + (size_t)48 * 1024, 1024), o3);
  }

#pragma unroll
  for (int r = 0; r < 8; ++r) {
    const float inv = 1.0f / lsum[r];
    const size_t orow =
        ((size_t)(b * ASEQ + qrow0 + r + 8 * hh)) * H_DIM + h * HEAD_D + l15;
    O[orow + 0]  = (bf16)(o0[r] * inv);
    O[orow + 16] = (bf16)(o1[r] * inv);
    O[orow + 32] = (bf16)(o2[r] * inv);
    O[orow + 48] = (bf16)(o3[r] * inv);
  }
}

// ---------------------------------------------------------------------------
// Fused (optional row-mask on `a`) + residual add + LayerNorm.
// One wave per row of 512. Writes f32 and/or bf16.
// ---------------------------------------------------------------------------
__global__ __launch_bounds__(32)
void add_layernorm(const float* __restrict__ a, const float* __restrict__ res,
                   const unsigned char* __restrict__ rowmask,
                   const float* __restrict__ g, const float* __restrict__ bb,
                   float* __restrict__ yf, bf16* __restrict__ yb)
{
  const int row = blockIdx.x;
  const int lane = threadIdx.x;
  const size_t base = (size_t)row * H_DIM;
  const bool keep = rowmask ? (rowmask[row] != 0) : true;
  float v[16];
  float s = 0.0f;
#pragma unroll
  for (int i = 0; i < 16; ++i) {
    const int c = lane + i * 32;
    float av = a[base + c];
    if (!keep) av = 0.0f;
    v[i] = av + res[base + c];
    s += v[i];
  }
#pragma unroll
  for (int o = 16; o >= 1; o >>= 1) s += __shfl_xor(s, o, 32);
  const float mu = s * (1.0f / 512.0f);
  float q = 0.0f;
#pragma unroll
  for (int i = 0; i < 16; ++i) { const float d = v[i] - mu; q += d * d; }
#pragma unroll
  for (int o = 16; o >= 1; o >>= 1) q += __shfl_xor(q, o, 32);
  const float rstd = rsqrtf(q * (1.0f / 512.0f) + 1e-5f);
#pragma unroll
  for (int i = 0; i < 16; ++i) {
    const int c = lane + i * 32;
    const float y = (v[i] - mu) * rstd * g[c] + bb[c];
    if (yf) yf[base + c] = y;
    if (yb) yb[base + c] = (bf16)y;
  }
}

// ---------------------------------------------------------------------------
// Host orchestration
// ---------------------------------------------------------------------------
extern "C" void kernel_launch(void* const* d_in, const int* in_sizes, int n_in,
                              void* d_out, int out_size, void* d_ws, size_t ws_size,
                              hipStream_t stream)
{
  (void)in_sizes; (void)n_in; (void)out_size; (void)ws_size;

  const float* x0        = (const float*)d_in[0];   // [8,1024,512]
  const float* vc        = (const float*)d_in[2];   // [1024,512]
  const int*   vbi       = (const int*)d_in[3];     // [1024]
  const unsigned char* tmask = (const unsigned char*)d_in[4]; // [8,1024] bool
  const float* sa_w_in   = (const float*)d_in[5];
  const float* sa_b_in   = (const float*)d_in[6];
  const float* sa_w_out  = (const float*)d_in[7];
  const float* sa_b_out  = (const float*)d_in[8];
  const float* ca_w_in   = (const float*)d_in[9];
  const float* ca_b_in   = (const float*)d_in[10];
  const float* ca_w_out  = (const float*)d_in[11];
  const float* ca_b_out  = (const float*)d_in[12];
  const float* ffn_w1    = (const float*)d_in[13];
  const float* ffn_b1    = (const float*)d_in[14];
  const float* ffn_w2    = (const float*)d_in[15];
  const float* ffn_b2    = (const float*)d_in[16];
  const float* ln1_g = (const float*)d_in[17];
  const float* ln1_b = (const float*)d_in[18];
  const float* ln2_g = (const float*)d_in[19];
  const float* ln2_b = (const float*)d_in[20];
  const float* ln3_g = (const float*)d_in[21];
  const float* ln3_b = (const float*)d_in[22];
  float* yout = (float*)d_out;

  char* ws = (char*)d_ws;
  const size_t MB = 1024ull * 1024ull;

  bf16* xb = (bf16*)(ws + 0 * MB);                 // 8 MB  [8192,512]
  size_t o = 8 * MB;                               // weight pool [8,16) MB
  bf16* wsain  = (bf16*)(ws + o); o += 1536 * 512 * 2;
  bf16* wsaout = (bf16*)(ws + o); o += 512 * 512 * 2;
  bf16* wcain  = (bf16*)(ws + o); o += 1536 * 512 * 2;
  bf16* wcaout = (bf16*)(ws + o); o += 512 * 512 * 2;
  bf16* wf1    = (bf16*)(ws + o); o += 1024 * 512 * 2;
  bf16* wf2    = (bf16*)(ws + o); o += 512 * 1024 * 2;
  bf16* vcb    = (bf16*)(ws + o); o += 1024 * 512 * 2;

  bf16* qkvb  = (bf16*)(ws + 16 * MB);   // 24 MB [8192,1536]
  bf16* caqb  = (bf16*)(ws + 16 * MB);   // alias (qkv dead after self-attn)
  bf16* cakvb = (bf16*)(ws + 24 * MB);   // 2 MB  [1024,1024]
  bf16* cavt  = (bf16*)(ws + 26 * MB);   // 1 MB  [8,64,1024]
  bf16* vt    = (bf16*)(ws + 40 * MB);   // 8 MB  [8,8,64,1024]
  bf16* attO  = (bf16*)(ws + 48 * MB);   // 8 MB  [8192,512]
  bf16* caO   = (bf16*)(ws + 48 * MB);   // alias (attO dead after proj)
  float* attp = (float*)(ws + 56 * MB);  // 16 MB [8192,512]
  bf16*  hb   = (bf16*) (ws + 56 * MB);  // alias (attp dead after LN1) [8192,1024]
  float* x1f  = (float*)(ws + 72 * MB);  // 16 MB
  bf16*  x1b  = (bf16*) (ws + 88 * MB);  // 8 MB
  float* cap  = (float*)(ws + 96 * MB);  // 16 MB
  float* h2   = (float*)(ws + 96 * MB);  // alias (cap dead after LN2)
  float* x2f  = (float*)(ws + 112 * MB); // 16 MB
  bf16*  x2b  = (bf16*) (ws + 128 * MB); // 8 MB

  auto cvt = [&](const float* src, bf16* dst, int n) {
    cvt_f32_bf16<<<(n + 255) / 256, 256, 0, stream>>>(src, dst, n);
  };
  auto gemm = [&](const bf16* A, int lda, const bf16* W, int ldw,
                  const float* bias, float* Cf, bf16* Cb, int ldc,
                  int M, int N, int K, int act) {
    dim3 g(N / 128, M / 128);
    gemm_bf16_wmma<<<g, 256, 0, stream>>>(A, lda, W, ldw, bias, Cf, Cb, ldc, K, act);
  };

  // --- stage 0: bf16 conversions ---
  cvt(x0,       xb,     8192 * 512);
  cvt(vc,       vcb,    1024 * 512);
  cvt(sa_w_in,  wsain,  1536 * 512);
  cvt(sa_w_out, wsaout, 512 * 512);
  cvt(ca_w_in,  wcain,  1536 * 512);
  cvt(ca_w_out, wcaout, 512 * 512);
  cvt(ffn_w1,   wf1,    1024 * 512);
  cvt(ffn_w2,   wf2,    512 * 1024);

  // --- self-attention ---
  gemm(xb, 512, wsain, 512, sa_b_in, nullptr, qkvb, 1536, 8192, 1536, 512, 0);
  {
    int n = BATCH * N_HEADS * HEAD_D * ASEQ;
    build_vt_self<<<(n + 255) / 256, 256, 0, stream>>>(qkvb, vt);
  }
  attn_wmma<<<dim3(BATCH * N_HEADS, ASEQ / 64), 128, 0, stream>>>(
      0, qkvb, qkvb, vt, tmask, vbi, attO);
  gemm(attO, 512, wsaout, 512, sa_b_out, attp, nullptr, 512, 8192, 512, 512, 0);
  add_layernorm<<<8192, 32, 0, stream>>>(attp, x0, nullptr, ln1_g, ln1_b, x1f, x1b);

  // --- cross-attention (atoms -> virtual nodes, per-batch membership mask) ---
  gemm(x1b, 512, wcain, 512, ca_b_in, nullptr, caqb, 512, 8192, 512, 512, 0);
  gemm(vcb, 512, wcain + (size_t)512 * 512, 512, ca_b_in + 512,
       nullptr, cakvb, 1024, 1024, 1024, 512, 0);
  {
    int n = N_HEADS * HEAD_D * NV;
    build_vt_cross<<<(n + 255) / 256, 256, 0, stream>>>(cakvb, cavt);
  }
  attn_wmma<<<dim3(BATCH * N_HEADS, ASEQ / 64), 128, 0, stream>>>(
      1, caqb, cakvb, cavt, tmask, vbi, caO);
  gemm(caO, 512, wcaout, 512, ca_b_out, cap, nullptr, 512, 8192, 512, 512, 0);
  add_layernorm<<<8192, 32, 0, stream>>>(cap, x1f, tmask, ln2_g, ln2_b, x2f, x2b);

  // --- FFN (exact GELU) ---
  gemm(x2b, 512, wf1, 512, ffn_b1, nullptr, hb, 1024, 8192, 1024, 512, 1);
  gemm(hb, 1024, wf2, 1024, ffn_b2, h2, nullptr, 512, 8192, 512, 1024, 0);
  add_layernorm<<<8192, 32, 0, stream>>>(h2, x2f, nullptr, ln3_g, ln3_b, yout, nullptr);
}
